// SchrodingerAttention_59485297050284
// MI455X (gfx1250) — matrix operand order repaired and checked by reading, not compiled, not run on hardware
//
#include <hip/hip_runtime.h>

#define NHEAD 16

typedef __bf16 v16bf __attribute__((ext_vector_type(16)));
typedef float  v8f   __attribute__((ext_vector_type(8)));
typedef unsigned int u32x4 __attribute__((ext_vector_type(4)));

union Frag16 { v16bf v; __bf16 h[16]; u32x4 q[2]; };
union AccF   { v8f  v; float  f[8]; };

__device__ __forceinline__ v8f wmma_bf16(const Frag16 &a, const Frag16 &b, v8f c) {
  return __builtin_amdgcn_wmma_f32_16x16x32_bf16(false, a.v, false, b.v, (short)0, c,
                                                 false, false);
}

// CDNA5 async global->LDS (ASYNCcnt tracked, no VGPR round trip).
__device__ __forceinline__ void async_load_b128(const void* g, void* lds) {
  unsigned l = (unsigned)(unsigned long long)lds;
  unsigned long long a = (unsigned long long)g;
  asm volatile("global_load_async_to_lds_b128 %0, %1, off" :: "v"(l), "v"(a) : "memory");
}
__device__ __forceinline__ void wait_async0() {
  asm volatile("s_wait_asynccnt 0" ::: "memory");
}

// ---------------- f32 -> bf16 convert ----------------
__global__ void k_f32_to_bf16(const float* __restrict__ in, __bf16* __restrict__ out, long n) {
  long i = (long)blockIdx.x * blockDim.x + threadIdx.x;
  long stride = (long)gridDim.x * blockDim.x;
  for (; i < n; i += stride) out[i] = (__bf16)in[i];
}

// ---------------- fused complex GEMM ----------------
// Cr = Ar@Wr^T - Ai@Wi^T ; Ci = Ar@Wi^T + Ai@Wr^T, with W = [Wr ; Wi] (2N x K).
// Block 256 thr (8 waves), tile 128x64, k-step 32, double-buffered async LDS.
// Wave owns 32x32 complex patch: 16 wmma accumulators, 16 WMMAs per k-step.
__global__ __launch_bounds__(256) void k_cgemm_bf16(
    const __bf16* __restrict__ Ar, const __bf16* __restrict__ Ai,
    const __bf16* __restrict__ W,
    float* __restrict__ Cr, float* __restrict__ Ci, int M, int N, int K) {
  __shared__ __align__(16) __bf16 Ars[2][128 * 40];
  __shared__ __align__(16) __bf16 Ais[2][128 * 40];
  __shared__ __align__(16) __bf16 Wrs[2][64 * 40];
  __shared__ __align__(16) __bf16 Wis[2][64 * 40];
  const int tid  = threadIdx.x;
  const int wid  = tid >> 5;
  const int lane = tid & 31;
  const int l16  = lane & 15;
  const int hi   = (lane >> 4) & 1;
  const int wm = wid & 3, wn = wid >> 2;
  const long mb = (long)blockIdx.y * 128;
  const long nb = (long)blockIdx.x * 64;
  const __bf16* Wr = W;
  const __bf16* Wi = W + (long)N * K;

  AccF rr[2][2], qq[2][2], ri[2][2], ir[2][2];
#pragma unroll
  for (int ms = 0; ms < 2; ++ms)
#pragma unroll
    for (int ns = 0; ns < 2; ++ns)
#pragma unroll
      for (int i = 0; i < 8; ++i) {
        rr[ms][ns].f[i] = 0.f; qq[ms][ns].f[i] = 0.f;
        ri[ms][ns].f[i] = 0.f; ir[ms][ns].f[i] = 0.f;
      }

  auto stage = [&](int buf, int kc) {
#pragma unroll
    for (int it = 0; it < 2; ++it) {
      int task = tid + it * 256;
      int row = task >> 2, g = task & 3;
      long off = (mb + row) * K + kc + g * 8;
      async_load_b128(&Ar[off], &Ars[buf][row * 40 + g * 8]);
      async_load_b128(&Ai[off], &Ais[buf][row * 40 + g * 8]);
    }
    {
      int row = tid >> 2, g = tid & 3;
      long off = (nb + row) * K + kc + g * 8;
      async_load_b128(&Wr[off], &Wrs[buf][row * 40 + g * 8]);
      async_load_b128(&Wi[off], &Wis[buf][row * 40 + g * 8]);
    }
  };

  int cur = 0;
  stage(0, 0);
  for (int kc = 0; kc < K; kc += 32) {
    wait_async0();
    __syncthreads();
    if (kc + 32 < K) stage(cur ^ 1, kc + 32);

    Frag16 ar[2], ai[2], br[2], bi[2];
#pragma unroll
    for (int ms = 0; ms < 2; ++ms) {
      const int am = wm * 32 + ms * 16 + l16, ab = hi * 8;
      ar[ms].q[0] = *(const u32x4*)&Ars[cur][am * 40 + ab];
      ar[ms].q[1] = *(const u32x4*)&Ars[cur][am * 40 + ab + 16];
      ai[ms].q[0] = *(const u32x4*)&Ais[cur][am * 40 + ab];
      ai[ms].q[1] = *(const u32x4*)&Ais[cur][am * 40 + ab + 16];
    }
#pragma unroll
    for (int ns = 0; ns < 2; ++ns) {
      const int n0 = wn * 32 + ns * 16 + l16, bb = hi * 8;
      br[ns].q[0] = *(const u32x4*)&Wrs[cur][n0 * 40 + bb];
      br[ns].q[1] = *(const u32x4*)&Wrs[cur][n0 * 40 + bb + 16];
      bi[ns].q[0] = *(const u32x4*)&Wis[cur][n0 * 40 + bb];
      bi[ns].q[1] = *(const u32x4*)&Wis[cur][n0 * 40 + bb + 16];
    }
#pragma unroll
    for (int ms = 0; ms < 2; ++ms)
#pragma unroll
      for (int ns = 0; ns < 2; ++ns) {
        rr[ms][ns].v = wmma_bf16(ar[ms], br[ns], rr[ms][ns].v);
        qq[ms][ns].v = wmma_bf16(ai[ms], bi[ns], qq[ms][ns].v);
        ri[ms][ns].v = wmma_bf16(ar[ms], bi[ns], ri[ms][ns].v);
        ir[ms][ns].v = wmma_bf16(ai[ms], br[ns], ir[ms][ns].v);
      }
    cur ^= 1;
  }

#pragma unroll
  for (int ms = 0; ms < 2; ++ms)
#pragma unroll
    for (int ns = 0; ns < 2; ++ns)
#pragma unroll
      for (int v = 0; v < 8; ++v) {
        long mm = mb + wm * 32 + ms * 16 + v + hi * 8;
        long nn = nb + wn * 32 + ns * 16 + l16;
        long idx = mm * (long)N + nn;
        Cr[idx] = rr[ms][ns].f[v] - qq[ms][ns].f[v];
        Ci[idx] = ri[ms][ns].f[v] + ir[ms][ns].f[v];
      }
}

// ---------------- real GEMM: C = A(MxK) @ W(NxK)^T ----------------
__global__ __launch_bounds__(256) void k_gemm_bf16(
    const __bf16* __restrict__ A, const __bf16* __restrict__ W,
    float* __restrict__ C, int M, int N, int K) {
  __shared__ __align__(16) __bf16 As[2][128 * 40];
  __shared__ __align__(16) __bf16 Ws[2][64 * 40];
  const int tid  = threadIdx.x;
  const int wid  = tid >> 5;
  const int lane = tid & 31;
  const int l16  = lane & 15;
  const int hi   = (lane >> 4) & 1;
  const int wm = wid & 3, wn = wid >> 2;
  const long mb = (long)blockIdx.y * 128;
  const long nb = (long)blockIdx.x * 64;

  AccF acc[2][2];
#pragma unroll
  for (int ms = 0; ms < 2; ++ms)
#pragma unroll
    for (int ns = 0; ns < 2; ++ns)
#pragma unroll
      for (int i = 0; i < 8; ++i) acc[ms][ns].f[i] = 0.f;

  auto stage = [&](int buf, int kc) {
#pragma unroll
    for (int it = 0; it < 2; ++it) {
      int task = tid + it * 256;
      int row = task >> 2, g = task & 3;
      async_load_b128(&A[(mb + row) * K + kc + g * 8], &As[buf][row * 40 + g * 8]);
    }
    {
      int row = tid >> 2, g = tid & 3;
      async_load_b128(&W[(nb + row) * K + kc + g * 8], &Ws[buf][row * 40 + g * 8]);
    }
  };

  int cur = 0;
  stage(0, 0);
  for (int kc = 0; kc < K; kc += 32) {
    wait_async0();
    __syncthreads();
    if (kc + 32 < K) stage(cur ^ 1, kc + 32);

    Frag16 af[2], bf[2];
#pragma unroll
    for (int ms = 0; ms < 2; ++ms) {
      const int am = wm * 32 + ms * 16 + l16, ab = hi * 8;
      af[ms].q[0] = *(const u32x4*)&As[cur][am * 40 + ab];
      af[ms].q[1] = *(const u32x4*)&As[cur][am * 40 + ab + 16];
    }
#pragma unroll
    for (int ns = 0; ns < 2; ++ns) {
      const int n0 = wn * 32 + ns * 16 + l16, bb = hi * 8;
      bf[ns].q[0] = *(const u32x4*)&Ws[cur][n0 * 40 + bb];
      bf[ns].q[1] = *(const u32x4*)&Ws[cur][n0 * 40 + bb + 16];
    }
#pragma unroll
    for (int ms = 0; ms < 2; ++ms)
#pragma unroll
      for (int ns = 0; ns < 2; ++ns)
        acc[ms][ns].v = wmma_bf16(af[ms], bf[ns], acc[ms][ns].v);
    cur ^= 1;
  }

#pragma unroll
  for (int ms = 0; ms < 2; ++ms)
#pragma unroll
    for (int ns = 0; ns < 2; ++ns)
#pragma unroll
      for (int v = 0; v < 8; ++v) {
        long mm = mb + wm * 32 + ms * 16 + v + hi * 8;
        long nn = nb + wn * 32 + ns * 16 + l16;
        C[mm * (long)N + nn] = acc[ms][ns].f[v];
      }
}

// ---------------- RoPE + complex RMS-norm, f32 in -> bf16 out ----------------
__global__ __launch_bounds__(128) void k_rope_norm(
    const float* __restrict__ xr, const float* __restrict__ xi,
    const float* __restrict__ cs, const float* __restrict__ sn,
    __bf16* __restrict__ obr, __bf16* __restrict__ obi, int T, int D) {
  const int wid = threadIdx.x >> 5, lane = threadIdx.x & 31;
  const int idx = blockIdx.x * 4 + wid;
  if (idx >= T * NHEAD) return;
  const int t = idx / NHEAD, hh = idx % NHEAD;
  const long base = (long)t * D + hh * 64;

  float ar0 = xr[base + lane],      ai0 = xi[base + lane];
  float ar1 = xr[base + 32 + lane], ai1 = xi[base + 32 + lane];
  float c = cs[t * 32 + lane], s = sn[t * 32 + lane];
  float nr0 = ar0 * c - ai0 * s;
  float ni0 = ar0 * s + ai0 * c;

  float sum = nr0 * nr0 + ni0 * ni0 + ar1 * ar1 + ai1 * ai1;
#pragma unroll
  for (int m = 1; m < 32; m <<= 1) sum += __shfl_xor(sum, m, 32);
  float mean = sum * (1.f / 64.f) + 1e-8f;
  float sc = rsqrtf(mean + 1.1920929e-7f);

  obr[base + lane]      = (__bf16)(nr0 * sc);
  obi[base + lane]      = (__bf16)(ni0 * sc);
  obr[base + 32 + lane] = (__bf16)(ar1 * sc);
  obi[base + 32 + lane] = (__bf16)(ai1 * sc);
}

// ---------------- windowed complex attention (flash-style, WMMA) ----------------
__global__ __launch_bounds__(128) void k_attn(
    const __bf16* __restrict__ qr, const __bf16* __restrict__ qi,
    const __bf16* __restrict__ kr, const __bf16* __restrict__ ki,
    const __bf16* __restrict__ vr, const __bf16* __restrict__ vi,
    __bf16* __restrict__ dzr, __bf16* __restrict__ dzi,
    const int* __restrict__ wlp, int T, int D) {
  const int h  = blockIdx.y;
  const int qb = blockIdx.x * 64;
  const int tid = threadIdx.x;
  const int wid = tid >> 5;
  const int lane = tid & 31;
  const int l16 = lane & 15;
  const int hi  = (lane >> 4) & 1;
  const int wl = *wlp;
  const bool win = (wl > 0 && wl < T);
  const int qw = qb + wid * 16;
  const float scale = rsqrtf((float)(D / NHEAD));

  __shared__ __align__(16) __bf16 Kls[2][32 * 80];
  __shared__ __align__(16) __bf16 Vls[2][64 * 40];
  __shared__ __align__(16) __bf16 Hls[4][3][16 * 32];

  Frag16 Qr[2], Qi[2];
  {
    const long trow = (long)(qw + l16) * D + h * 64;
#pragma unroll
    for (int c = 0; c < 2; ++c)
#pragma unroll
      for (int j = 0; j < 16; ++j) {
        int d = c * 32 + hi * 8 + (j < 8 ? j : j + 8);
        Qr[c].h[j] = qr[trow + d];
        Qi[c].h[j] = qi[trow + d];
      }
  }

  float mM[8], lL[8];
  AccF ozr[4], ozi[4];
#pragma unroll
  for (int v = 0; v < 8; ++v) { mM[v] = 1e-4f; lL[v] = 0.f; }
#pragma unroll
  for (int s = 0; s < 4; ++s)
#pragma unroll
    for (int i = 0; i < 8; ++i) { ozr[s].f[i] = 0.f; ozi[s].f[i] = 0.f; }

  int lo  = win ? max(0, qw - wl + 1) : 0;
  int kw0 = lo & ~31;
  int kw1 = ((qw + 16 + 31) >> 5) << 5; if (kw1 > T) kw1 = T;
  const int nvis = kw1 - kw0;
  int lob = win ? max(0, qb - wl + 1) : 0;
  int kS = lob & ~31;
  int kE = qb + 64; if (kE > T) kE = T;

  for (int kt = kS; kt < kE; kt += 32) {
    for (int task = tid; task < 256; task += 128) {
      int row = task >> 3, g = task & 7;
      long src = (long)(kt + row) * D + h * 64 + g * 8;
      async_load_b128(&kr[src], &Kls[0][row * 80 + g * 8]);
      async_load_b128(&ki[src], &Kls[1][row * 80 + g * 8]);
      u32x4 a4 = *(const u32x4*)&vr[src];
      u32x4 b4 = *(const u32x4*)&vi[src];
      const __bf16* ah = (const __bf16*)&a4;
      const __bf16* bh = (const __bf16*)&b4;
#pragma unroll
      for (int e = 0; e < 8; ++e) {
        Vls[0][(g * 8 + e) * 40 + row] = ah[e];
        Vls[1][(g * 8 + e) * 40 + row] = bh[e];
      }
    }
    wait_async0();
    __syncthreads();

    if (kt >= kw0 && kt < kw1) {
      float sr[2][8], si[2][8], sab[2][8];
#pragma unroll
      for (int s = 0; s < 2; ++s) {
        Frag16 KrB[2], KiB[2];
#pragma unroll
        for (int c = 0; c < 2; ++c) {
          int off = (s * 16 + l16) * 80 + c * 32 + hi * 8;
          KrB[c].q[0] = *(const u32x4*)&Kls[0][off];
          KrB[c].q[1] = *(const u32x4*)&Kls[0][off + 16];
          KiB[c].q[0] = *(const u32x4*)&Kls[1][off];
          KiB[c].q[1] = *(const u32x4*)&Kls[1][off + 16];
        }
        AccF rr, ii, ir, ri;
#pragma unroll
        for (int i = 0; i < 8; ++i) { rr.f[i]=0.f; ii.f[i]=0.f; ir.f[i]=0.f; ri.f[i]=0.f; }
        rr.v = wmma_bf16(Qr[0], KrB[0], rr.v); rr.v = wmma_bf16(Qr[1], KrB[1], rr.v);
        ii.v = wmma_bf16(Qi[0], KiB[0], ii.v); ii.v = wmma_bf16(Qi[1], KiB[1], ii.v);
        ir.v = wmma_bf16(Qi[0], KrB[0], ir.v); ir.v = wmma_bf16(Qi[1], KrB[1], ir.v);
        ri.v = wmma_bf16(Qr[0], KiB[0], ri.v); ri.v = wmma_bf16(Qr[1], KiB[1], ri.v);
#pragma unroll
        for (int v = 0; v < 8; ++v) {
          int q = qw + v + hi * 8;
          int k = kt + s * 16 + l16;
          bool valid = (k <= q) && (!win || (q - k) <= wl - 1);
          float a = valid ? (rr.f[v] + ii.f[v]) * scale : 0.f;
          float b = valid ? (ir.f[v] - ri.f[v]) * scale : 0.f;
          sr[s][v] = a; si[s][v] = b;
          sab[s][v] = sqrtf(a * a + b * b + 1e-8f);
        }
      }

      float corr[8];
#pragma unroll
      for (int v = 0; v < 8; ++v) {
        float rm = fmaxf(sab[0][v], sab[1][v]);
#pragma unroll
        for (int m = 1; m < 16; m <<= 1) rm = fmaxf(rm, __shfl_xor(rm, m, 16));
        float mn = fmaxf(mM[v], rm);
        corr[v] = __expf(mM[v] - mn);
        mM[v] = mn;
        float e0 = __expf(sab[0][v] - mn);
        float e1 = __expf(sab[1][v] - mn);
        float rs = e0 + e1;
#pragma unroll
        for (int m = 1; m < 16; m <<= 1) rs += __shfl_xor(rs, m, 16);
        lL[v] = lL[v] * corr[v] + rs;

        int rowm = (v + hi * 8) * 32;
        float hr0 = sr[0][v] * e0, hr1 = sr[1][v] * e1;
        float hi0 = si[0][v] * e0, hi1 = si[1][v] * e1;
        Hls[wid][0][rowm + l16]      = (__bf16)hr0;
        Hls[wid][0][rowm + 16 + l16] = (__bf16)hr1;
        Hls[wid][1][rowm + l16]      = (__bf16)hi0;
        Hls[wid][1][rowm + 16 + l16] = (__bf16)hi1;
        Hls[wid][2][rowm + l16]      = (__bf16)(-hr0);
        Hls[wid][2][rowm + 16 + l16] = (__bf16)(-hr1);
      }
#pragma unroll
      for (int s = 0; s < 4; ++s)
#pragma unroll
        for (int v = 0; v < 8; ++v) { ozr[s].f[v] *= corr[v]; ozi[s].f[v] *= corr[v]; }

      __builtin_amdgcn_wave_barrier();
      asm volatile("s_wait_dscnt 0" ::: "memory");

      Frag16 HrA, HiA, HrnA;
      {
        const int ab = hi * 8;
        HrA.q[0]  = *(const u32x4*)&Hls[wid][0][l16 * 32 + ab];
        HrA.q[1]  = *(const u32x4*)&Hls[wid][0][l16 * 32 + ab + 16];
        HiA.q[0]  = *(const u32x4*)&Hls[wid][1][l16 * 32 + ab];
        HiA.q[1]  = *(const u32x4*)&Hls[wid][1][l16 * 32 + ab + 16];
        HrnA.q[0] = *(const u32x4*)&Hls[wid][2][l16 * 32 + ab];
        HrnA.q[1] = *(const u32x4*)&Hls[wid][2][l16 * 32 + ab + 16];
      }
#pragma unroll
      for (int ds = 0; ds < 4; ++ds) {
        Frag16 VrB, ViB;
        int off = (ds * 16 + l16) * 40 + hi * 8;
        VrB.q[0] = *(const u32x4*)&Vls[0][off];
        VrB.q[1] = *(const u32x4*)&Vls[0][off + 16];
        ViB.q[0] = *(const u32x4*)&Vls[1][off];
        ViB.q[1] = *(const u32x4*)&Vls[1][off + 16];
        ozr[ds].v = wmma_bf16(HiA,  VrB, ozr[ds].v);
        ozr[ds].v = wmma_bf16(HrA,  ViB, ozr[ds].v);
        ozi[ds].v = wmma_bf16(HiA,  ViB, ozi[ds].v);
        ozi[ds].v = wmma_bf16(HrnA, VrB, ozi[ds].v);
      }
    }
    __syncthreads();
  }

#pragma unroll
  for (int v = 0; v < 8; ++v) {
    lL[v] += (float)(T - nvis) * __expf(1e-4f - mM[v]);
    float inv = 1.f / lL[v];
    long trow = (long)(qw + v + hi * 8) * D + h * 64;
#pragma unroll
    for (int ds = 0; ds < 4; ++ds) {
      dzr[trow + ds * 16 + l16] = (__bf16)(ozr[ds].f[v] * inv);
      dzi[trow + ds * 16 + l16] = (__bf16)(ozi[ds].f[v] * inv);
    }
  }
}

// ---------------- host ----------------
extern "C" void kernel_launch(void* const* d_in, const int* in_sizes, int n_in,
                              void* d_out, int out_size, void* d_ws, size_t ws_size,
                              hipStream_t stream) {
  const float* xr = (const float*)d_in[0];
  const float* xi = (const float*)d_in[1];
  const float* cs = (const float*)d_in[2];
  const float* sn = (const float*)d_in[3];
  const float* wq = (const float*)d_in[4];
  const float* wk = (const float*)d_in[5];
  const float* wv = (const float*)d_in[6];
  const float* wp = (const float*)d_in[7];
  const int*   wl = (const int*)d_in[8];

  int D = 1; while ((long)D * D < (long)in_sizes[7]) ++D;
  const int T = in_sizes[0] / D;
  const long TD = (long)T * D;
  const long WW = 2L * D * D;

  char* p = (char*)d_ws;
  auto carve = [&](long bytes) { char* r = p; p += (bytes + 255) & ~255L; return r; };
  __bf16* xr_b = (__bf16*)carve(TD * 2);
  __bf16* xi_b = (__bf16*)carve(TD * 2);
  __bf16* wq_b = (__bf16*)carve(WW * 2);
  __bf16* wk_b = (__bf16*)carve(WW * 2);
  __bf16* wv_b = (__bf16*)carve(WW * 2);
  __bf16* wp_b = (__bf16*)carve((long)D * D * 2);
  float*  qrf  = (float*)carve(TD * 4);
  float*  qif  = (float*)carve(TD * 4);
  float*  krf  = (float*)carve(TD * 4);
  float*  kif  = (float*)carve(TD * 4);
  float*  vrf  = (float*)carve(TD * 4);
  float*  vif  = (float*)carve(TD * 4);
  __bf16* qbr  = (__bf16*)carve(TD * 2);
  __bf16* qbi  = (__bf16*)carve(TD * 2);
  __bf16* kbr  = (__bf16*)carve(TD * 2);
  __bf16* kbi  = (__bf16*)carve(TD * 2);
  __bf16* vbr  = (__bf16*)carve(TD * 2);
  __bf16* vbi  = (__bf16*)carve(TD * 2);
  __bf16* dzr  = (__bf16*)carve(TD * 2);
  __bf16* dzi  = (__bf16*)carve(TD * 2);

  auto cvt = [&](const float* s, __bf16* d, long n) {
    unsigned blocks = (unsigned)((n + 1023) / 1024);
    k_f32_to_bf16<<<blocks, 256, 0, stream>>>(s, d, n);
  };
  auto cgemm = [&](const __bf16* Wc, float* Cr, float* Ci) {
    dim3 grid(D / 64, T / 128);
    k_cgemm_bf16<<<grid, 256, 0, stream>>>(xr_b, xi_b, Wc, Cr, Ci, T, D, D);
  };

  cvt(xr, xr_b, TD); cvt(xi, xi_b, TD);
  cvt(wq, wq_b, WW); cvt(wk, wk_b, WW); cvt(wv, wv_b, WW);
  cvt(wp, wp_b, (long)D * D);

  // fused complex QKV projections (one launch each)
  cgemm(wq_b, qrf, qif);
  cgemm(wk_b, krf, kif);
  cgemm(wv_b, vrf, vif);

  {
    unsigned blocks = (unsigned)((T * NHEAD + 3) / 4);
    k_rope_norm<<<blocks, 128, 0, stream>>>(qrf, qif, cs, sn, qbr, qbi, T, D);
    k_rope_norm<<<blocks, 128, 0, stream>>>(krf, kif, cs, sn, kbr, kbi, T, D);
  }
  cvt(vrf, vbr, TD); cvt(vif, vbi, TD);

  {
    dim3 grid(T / 64, NHEAD);
    k_attn<<<grid, 128, 0, stream>>>(qbr, qbi, kbr, kbi, vbr, vbi, dzr, dzi, wl, T, D);
  }

  // output projection straight into d_out: [out_r | out_i]
  float* outp = (float*)d_out;
  dim3 grid(D / 64, T / 128);
  k_gemm_bf16<<<grid, 256, 0, stream>>>(dzr, wp_b, outp,      T, D, D);
  k_gemm_bf16<<<grid, 256, 0, stream>>>(dzi, wp_b, outp + TD, T, D, D);
}
